// MultiHeadAttention_9878424780806
// MI455X (gfx1250) — matrix operand order repaired and checked by reading, not compiled, not run on hardware
//
#include <hip/hip_runtime.h>
#include <hip/hip_bf16.h>

#define B_  4
#define S_  2048
#define D_  1024
#define H_  16
#define HD_ 64

typedef __bf16 bf16_t;
typedef __attribute__((ext_vector_type(16))) __bf16 v16bf;
typedef __attribute__((ext_vector_type(8)))  __bf16 v8bf;
typedef __attribute__((ext_vector_type(8)))  float  v8f;

union F16U { v16bf v; v8bf h[2]; };
union BFU  { unsigned short s; bf16_t b; };
union U4   { uint4 u; bf16_t b[8]; };

__device__ __forceinline__ bf16_t f2bf(float f) {
  union { float f; unsigned u; } a; a.f = f;
  unsigned r = a.u + 0x7FFFu + ((a.u >> 16) & 1u);   // round-to-nearest-even
  BFU o; o.s = (unsigned short)(r >> 16);
  return o.b;
}

// CDNA5 async copy: global -> LDS, 16B per lane, tracked by ASYNCcnt.
__device__ __forceinline__ void async_copy_b128(unsigned lds_off, const void* gptr) {
  asm volatile("global_load_async_to_lds_b128 %0, %1, off"
               :: "v"(lds_off), "v"(gptr) : "memory");
}
__device__ __forceinline__ void wait_async0() {
  asm volatile("s_wait_asynccnt 0x0" ::: "memory");
}

// A-fragment (16x32, MxK): lane m = l&15, half h = l>>4.
// elements 0..7  -> k = koff + h*8 + (0..7)
// elements 8..15 -> k = koff + 16 + h*8 + (0..7)
__device__ __forceinline__ v16bf load_frag_a(const bf16_t* base, int stride, int koff) {
  const int l = threadIdx.x & 31;
  const bf16_t* p = base + (l & 15) * stride + koff + (l >> 4) * 8;
  F16U f;
  f.h[0] = *(const v8bf*)(p);
  f.h[1] = *(const v8bf*)(p + 16);
  return f.v;
}

// B-fragment (32x16, KxN) from an LDS tile stored transposed [N][K]:
// lane n = l&15, half h = l>>4, elements 0..7 -> k = koff + h*8 + e, 8..15 -> k = koff + 16 + h*8 + (e-8)
__device__ __forceinline__ v16bf load_frag_b(const bf16_t* base, int stride, int koff) {
  const int l = threadIdx.x & 31;
  const bf16_t* p = base + (l & 15) * stride + koff + (l >> 4) * 8;
  F16U f;
  f.h[0] = ((const v8bf*)p)[0];
  f.h[1] = *(const v8bf*)(p + 16);
  return f.v;
}

__device__ __forceinline__ v8f wmma_bf16(v16bf a, v16bf b, v8f c) {
  return __builtin_amdgcn_wmma_f32_16x16x32_bf16(false, a, false, b, (short)0, c, false, false);
}

__global__ void cvt_bf16(const float* __restrict__ src, bf16_t* __restrict__ dst, int n) {
  int i = blockIdx.x * 256 + threadIdx.x;
  if (i < n) dst[i] = f2bf(src[i]);
}

// C[M,N] = A[M,K](bf16) @ B[K,N](bf16) + bias, BM=BN=128, BK=64, double-buffered.
// MODE 0: scatter into q/k/v [B,H,S,HD] bf16.  MODE 1: f32 row-major out.
template <int MODE>
__global__ __launch_bounds__(256) void gemm_wmma(
    const bf16_t* __restrict__ A, const bf16_t* __restrict__ Bm,
    const float* __restrict__ bias,
    bf16_t* __restrict__ qd, bf16_t* __restrict__ kd, bf16_t* __restrict__ vd,
    float* __restrict__ outf, int M, int N, int K)
{
  __shared__ bf16_t As[2][128 * 72];   // [row][k], stride 72 (144B): b128-aligned, conflict-free
  __shared__ bf16_t Bs[2][128 * 72];   // [n][k] (transposed)

  const int t = threadIdx.x;
  const int w  = t >> 5;
  const int wm = w >> 2;            // 0..1 : 64 rows each
  const int wn = w & 3;             // 0..3 : 32 cols each
  const int rowBase = blockIdx.y * 128;
  const int colBase = blockIdx.x * 128;

  v8f zero = {};
  v8f acc[4][2];
  #pragma unroll
  for (int mt = 0; mt < 4; mt++)
    #pragma unroll
    for (int nt = 0; nt < 2; nt++) acc[mt][nt] = zero;

  // issue async A-tile copies (128x64) into buffer `buf`
  auto stage_a = [&](int buf, int kb) {
    const unsigned base = (unsigned)(uintptr_t)(&As[buf][0]);
    #pragma unroll
    for (int i = 0; i < 4; i++) {
      int u = t + i * 256;              // 0..1023
      int row = u >> 3;
      int dc  = (u & 7) * 8;
      async_copy_b128(base + (unsigned)(row * 72 + dc) * 2,
                      A + (size_t)(rowBase + row) * K + kb + dc);
    }
  };
  // B tile 64x128: global loads held in VGPRs (register pipeline stage)
  auto load_b = [&](int kb, U4* br) {
    #pragma unroll
    for (int i = 0; i < 4; i++) {
      int u = t + i * 256;
      int kr = u >> 4;                  // 0..63
      int nc = (u & 15) * 8;
      br[i].u = *(const uint4*)(Bm + (size_t)(kb + kr) * N + colBase + nc);
    }
  };
  // transposed scatter of the held B tile into buffer `buf`
  auto store_b = [&](int buf, const U4* br) {
    #pragma unroll
    for (int i = 0; i < 4; i++) {
      int u = t + i * 256;
      int kr = u >> 4;
      int nc = (u & 15) * 8;
      #pragma unroll
      for (int e = 0; e < 8; e++) Bs[buf][(nc + e) * 72 + kr] = br[i].b[e];
    }
  };

  U4 breg[4];
  load_b(0, breg);
  stage_a(0, 0);

  int cur = 0;
  for (int kb = 0; kb < K; kb += 64) {
    wait_async0();                      // A copies issued last iteration -> As[cur] complete
    store_b(cur, breg);                 // B tile loaded last iteration -> Bs[cur]
    __syncthreads();
    if (kb + 64 < K) {                  // overlap next tile's traffic with this compute
      load_b(kb + 64, breg);
      stage_a(cur ^ 1, kb + 64);
    }
    #pragma unroll
    for (int kk = 0; kk < 64; kk += 32) {
      v16bf af[4], bfr[2];
      #pragma unroll
      for (int mt = 0; mt < 4; mt++)
        af[mt] = load_frag_a(&As[cur][(wm * 64 + mt * 16) * 72], 72, kk);
      #pragma unroll
      for (int nt = 0; nt < 2; nt++)
        bfr[nt] = load_frag_b(&Bs[cur][(wn * 32 + nt * 16) * 72], 72, kk);
      #pragma unroll
      for (int mt = 0; mt < 4; mt++)
        #pragma unroll
        for (int nt = 0; nt < 2; nt++)
          acc[mt][nt] = wmma_bf16(af[mt], bfr[nt], acc[mt][nt]);
    }
    cur ^= 1;
  }

  // epilogue: C element r -> (m = r + 8*(lane>>4), n = lane&15)
  const int l = t & 31, n = l & 15, hf = l >> 4;
  #pragma unroll
  for (int mt = 0; mt < 4; mt++) {
    #pragma unroll
    for (int nt = 0; nt < 2; nt++) {
      const int col0 = colBase + wn * 32 + nt * 16;
      const int col  = col0 + n;
      const float bv = bias[col];
      #pragma unroll
      for (int r = 0; r < 8; r++) {
        const int R = rowBase + wm * 64 + mt * 16 + r + 8 * hf;
        const float val = acc[mt][nt][r] + bv;
        if (MODE == 0) {
          const int which = col >> 10;
          const int rem = col & 1023;
          const int hh = rem >> 6, hd = rem & 63;
          const int bb = R >> 11, ss = R & 2047;
          const size_t idx = (((size_t)bb * H_ + hh) * S_ + ss) * HD_ + hd;
          const bf16_t bvb = f2bf(val);
          if (which == 0) qd[idx] = bvb;
          else if (which == 1) kd[idx] = bvb;
          else vd[idx] = bvb;
        } else {
          outf[(size_t)R * N + col] = val;
        }
      }
    }
  }
}

// Flash attention: block = 8 waves, 128 q rows; 32-key chunks, double-buffered K/V.
__global__ __launch_bounds__(256) void attn_wmma(
    const bf16_t* __restrict__ q, const bf16_t* __restrict__ k,
    const bf16_t* __restrict__ v, const float* __restrict__ mask,
    bf16_t* __restrict__ outb)
{
  __shared__ bf16_t kt[2][32 * 72];     // [key][d], stride 72
  __shared__ bf16_t vt[2][64 * 40];     // [d][key] transposed, stride 40
  __shared__ bf16_t pt[8 * 16 * 32];    // per-wave P tile (C-layout -> A-layout round trip)

  const int t = threadIdx.x;
  const int w = t >> 5;
  const int l = t & 31, n = l & 15, hf = l >> 4;
  const int bh = blockIdx.y;
  const int bb = bh >> 4, hh = bh & 15;
  const int q0 = blockIdx.x * 128 + w * 16;

  const bf16_t* qp = q + ((size_t)bh * S_ + q0) * HD_;
  const bf16_t* kp = k + (size_t)bh * S_ * HD_;
  const bf16_t* vp = v + (size_t)bh * S_ * HD_;

  const int key = t >> 3, dc = (t & 7) * 8;   // cooperative-staging coordinates

  auto stage_k = [&](int buf, int jb) {
    const unsigned base = (unsigned)(uintptr_t)(&kt[buf][0]);
    async_copy_b128(base + (unsigned)(key * 72 + dc) * 2,
                    kp + (size_t)(jb + key) * HD_ + dc);
  };
  auto load_v = [&](int jb, U4& vr) {
    vr.u = *(const uint4*)(vp + (size_t)(jb + key) * HD_ + dc);
  };
  auto store_v = [&](int buf, const U4& vr) {
    #pragma unroll
    for (int e = 0; e < 8; e++) vt[buf][(dc + e) * 40 + key] = vr.b[e];
  };

  // Q fragments held in registers for the whole pass
  const v16bf qa0 = load_frag_a(qp, HD_, 0);
  const v16bf qa1 = load_frag_a(qp, HD_, 32);

  v8f zero = {};
  v8f o[4];
  #pragma unroll
  for (int nt = 0; nt < 4; nt++) o[nt] = zero;
  float rmax[8], rsum[8];
  #pragma unroll
  for (int r = 0; r < 8; r++) { rmax[r] = -1e30f; rsum[r] = 0.0f; }

  bf16_t* pw = pt + w * (16 * 32);
  const float scale = 0.125f;           // 1/sqrt(64)

  U4 vreg;
  load_v(0, vreg);
  stage_k(0, 0);

  int cur = 0;
  for (int jb = 0; jb < S_; jb += 32) {
    wait_async0();                      // K copies for kt[cur] complete
    store_v(cur, vreg);                 // V chunk held since last iteration
    __syncthreads();
    if (jb + 32 < S_) {                 // overlap next chunk's traffic with compute
      load_v(jb + 32, vreg);
      stage_k(cur ^ 1, jb + 32);
    }

    // scores: two 16x16 tiles over keys [jb, jb+32)
    v8f s0 = zero, s1 = zero;
    s0 = wmma_bf16(qa0, load_frag_b(&kt[cur][0], 72, 0),  s0);
    s0 = wmma_bf16(qa1, load_frag_b(&kt[cur][0], 72, 32), s0);
    s1 = wmma_bf16(qa0, load_frag_b(&kt[cur][16 * 72], 72, 0),  s1);
    s1 = wmma_bf16(qa1, load_frag_b(&kt[cur][16 * 72], 72, 32), s1);

    // online softmax per row (rows split across halves; reduce within 16-lane group)
    #pragma unroll
    for (int r = 0; r < 8; r++) {
      const int qs = q0 + r + 8 * hf;
      const float m0 = mask[(size_t)qs * S_ + jb + n];
      const float m1 = mask[(size_t)qs * S_ + jb + 16 + n];
      float x0 = s0[r] * scale + m0;
      float x1 = s1[r] * scale + m1;
      float mx = fmaxf(x0, x1);
      mx = fmaxf(mx, __shfl_xor(mx, 1));
      mx = fmaxf(mx, __shfl_xor(mx, 2));
      mx = fmaxf(mx, __shfl_xor(mx, 4));
      mx = fmaxf(mx, __shfl_xor(mx, 8));
      const float nm = fmaxf(rmax[r], mx);
      const float corr = __expf(rmax[r] - nm);
      const float p0 = __expf(x0 - nm);
      const float p1 = __expf(x1 - nm);
      float rs = p0 + p1;
      rs += __shfl_xor(rs, 1);
      rs += __shfl_xor(rs, 2);
      rs += __shfl_xor(rs, 4);
      rs += __shfl_xor(rs, 8);
      rsum[r] = rsum[r] * corr + rs;
      rmax[r] = nm;
      #pragma unroll
      for (int nt = 0; nt < 4; nt++) o[nt][r] = o[nt][r] * corr;
      const int m = r + 8 * hf;
      pw[m * 32 + n]      = f2bf(p0);
      pw[m * 32 + 16 + n] = f2bf(p1);
    }

    // O += P @ V  (P as A-frag from LDS, V transposed B-frags)
    const v16bf pa = load_frag_a(pw, 32, 0);
    #pragma unroll
    for (int nt = 0; nt < 4; nt++)
      o[nt] = wmma_bf16(pa, load_frag_b(&vt[cur][nt * 16 * 40], 40, 0), o[nt]);

    cur ^= 1;
  }

  // normalize + store to [B,S,D] bf16 for the output GEMM
  #pragma unroll
  for (int nt = 0; nt < 4; nt++) {
    #pragma unroll
    for (int r = 0; r < 8; r++) {
      const int m = r + 8 * hf;
      const int ss = q0 + m;
      const float val = o[nt][r] / rsum[r];
      outb[((size_t)bb * S_ + ss) * D_ + hh * HD_ + nt * 16 + n] = f2bf(val);
    }
  }
}

extern "C" void kernel_launch(void* const* d_in, const int* in_sizes, int n_in,
                              void* d_out, int out_size, void* d_ws, size_t ws_size,
                              hipStream_t stream) {
  const float* x    = (const float*)d_in[0];
  const float* mask = (const float*)d_in[1];
  const float* Wqkv = (const float*)d_in[2];
  const float* bqkv = (const float*)d_in[3];
  const float* Wout = (const float*)d_in[4];
  const float* bout = (const float*)d_in[5];
  float* out = (float*)d_out;

  char* ws = (char*)d_ws;
  size_t off = 0;
  auto alloc = [&](size_t bytes) -> char* {
    char* p = ws + off;
    off += (bytes + 255) & ~(size_t)255;
    return p;
  };
  bf16_t* xb    = (bf16_t*)alloc((size_t)B_ * S_ * D_ * 2);
  bf16_t* wqkvb = (bf16_t*)alloc((size_t)D_ * 3 * D_ * 2);
  bf16_t* woutb = (bf16_t*)alloc((size_t)D_ * D_ * 2);
  bf16_t* qb    = (bf16_t*)alloc((size_t)B_ * H_ * S_ * HD_ * 2);
  bf16_t* kb    = (bf16_t*)alloc((size_t)B_ * H_ * S_ * HD_ * 2);
  bf16_t* vb    = (bf16_t*)alloc((size_t)B_ * H_ * S_ * HD_ * 2);
  bf16_t* ab    = (bf16_t*)alloc((size_t)B_ * S_ * D_ * 2);

  const int n1 = B_ * S_ * D_;
  cvt_bf16<<<(n1 + 255) / 256, 256, 0, stream>>>(x, xb, n1);
  const int n2 = D_ * 3 * D_;
  cvt_bf16<<<(n2 + 255) / 256, 256, 0, stream>>>(Wqkv, wqkvb, n2);
  const int n3 = D_ * D_;
  cvt_bf16<<<(n3 + 255) / 256, 256, 0, stream>>>(Wout, woutb, n3);

  dim3 g1(3 * D_ / 128, (B_ * S_) / 128);   // (24, 64)
  gemm_wmma<0><<<g1, 256, 0, stream>>>(xb, wqkvb, bqkv, qb, kb, vb, nullptr,
                                       B_ * S_, 3 * D_, D_);

  dim3 g2(S_ / 128, B_ * H_);               // (16, 64)
  attn_wmma<<<g2, 256, 0, stream>>>(qb, kb, vb, mask, ab);

  dim3 g3(D_ / 128, (B_ * S_) / 128);       // (8, 64)
  gemm_wmma<1><<<g3, 256, 0, stream>>>(ab, woutb, bout, nullptr, nullptr, nullptr,
                                       out, B_ * S_, D_, D_);
}
